// VisualMemoryScorer_83253646065894
// MI455X (gfx1250) — matrix operand order repaired and checked by reading, not compiled, not run on hardware
//
#include <hip/hip_runtime.h>

typedef __bf16 bf16_t;
typedef __attribute__((ext_vector_type(16))) __bf16 v16bf;
typedef __attribute__((ext_vector_type(8)))  __bf16 v8bf;
typedef __attribute__((ext_vector_type(8)))  float  v8f;
typedef int v4i_gcc __attribute__((vector_size(16)));   // matches builtin param type

#if defined(__HIP_DEVICE_COMPILE__) && __has_builtin(__builtin_amdgcn_global_load_async_to_lds_b128)
#define ASYNC_LDS 1
#else
#define ASYNC_LDS 0
#endif

__device__ __forceinline__ float sigmoidf_(float x) { return 1.0f / (1.0f + expf(-x)); }
__device__ __forceinline__ float gelu_exact(float x) { return 0.5f * x * (1.0f + erff(x * 0.70710678118654752f)); }

// ---------------------------------------------------------------------------
// WMMA fragment loads (ISA 7.12.2, 16-bit operands, wave32).
// A 16x32 fragment: lane m = lane&15; kb = (lane>=16)?8:0;
//   elems 0..7  <- k = kb..kb+7 ; elems 8..15 <- k = 16+kb..16+kb+7
// B 32x16 fragment: lane n = lane&15; kb = (lane>=16)?8:0;
//   elems 0..7  <- k = kb..kb+7 ; elems 8..15 <- k = 16+kb..16+kb+7
// Both reduce to aligned 16-byte vector loads.
// ---------------------------------------------------------------------------
__device__ __forceinline__ v16bf load_frag_a(const bf16_t* __restrict__ row, int lane) {
    int kb = ((lane >> 4) << 3);
    v8bf lo = *(const v8bf*)(row + kb);
    v8bf hi = *(const v8bf*)(row + 16 + kb);
    return __builtin_shufflevector(lo, hi, 0, 1, 2, 3, 4, 5, 6, 7,
                                           8, 9, 10, 11, 12, 13, 14, 15);
}
__device__ __forceinline__ v16bf load_frag_b(const bf16_t* __restrict__ row, int lane) {
    int kb = ((lane >> 4) << 3);
    v8bf lo = *(const v8bf*)(row + kb);
    v8bf hi = *(const v8bf*)(row + 16 + kb);
    return __builtin_shufflevector(lo, hi, 0, 1, 2, 3, 4, 5, 6, 7,
                                           8, 9, 10, 11, 12, 13, 14, 15);
}

// ---------------------------------------------------------------------------
// Async global->LDS staging (CDNA5 GLOBAL_LOAD_ASYNC_TO_LDS_B128, ASYNCcnt),
// with a synchronous uint4 fallback when the builtin is unavailable.
// ---------------------------------------------------------------------------
__device__ __forceinline__ void stage16(const bf16_t* __restrict__ g, bf16_t* l) {
#if ASYNC_LDS
    __builtin_amdgcn_global_load_async_to_lds_b128(
        (__attribute__((address_space(1))) v4i_gcc*)(uintptr_t)g,
        (__attribute__((address_space(3))) v4i_gcc*)(unsigned)(uintptr_t)l, 0, 0);
#else
    *(uint4*)l = *(const uint4*)g;
#endif
}
__device__ __forceinline__ void stage_wait() {
#if ASYNC_LDS
#if __has_builtin(__builtin_amdgcn_s_wait_asynccnt)
    __builtin_amdgcn_s_wait_asynccnt(0);
#else
    asm volatile("s_wait_asynccnt 0" ::: "memory");
#endif
#endif
}

// ---------------------------------------------------------------------------
// f32 -> bf16 (optionally strided sub-matrix) conversion
// ---------------------------------------------------------------------------
__global__ __launch_bounds__(256) void cvt_kernel(const float* __restrict__ src,
                                                  bf16_t* __restrict__ dst,
                                                  int rows, int cols, int ld, int off) {
    int idx = blockIdx.x * 256 + threadIdx.x;
    if (idx >= rows * cols) return;
    int r = idx / cols, c = idx - r * cols;
    dst[idx] = (bf16_t)src[(size_t)r * ld + off + c];
}

// features [B,D,T] f32  ->  X [B*T, D] bf16
__global__ __launch_bounds__(256) void tin_kernel(const float* __restrict__ feat,
                                                  bf16_t* __restrict__ X) {
    size_t idx = (size_t)blockIdx.x * 256 + threadIdx.x;   // ordered (b,d,t)
    if (idx >= (size_t)16 * 1024 * 1024) return;
    int t = (int)(idx & 1023);
    size_t bd = idx >> 10;
    int d = (int)(bd & 1023);
    int b = (int)(bd >> 10);
    X[(((size_t)b << 10) + t) * 1024 + d] = (bf16_t)feat[idx];
}

// fused [B*T, 256] f32 -> frame [B, 256, T] f32
__global__ __launch_bounds__(256) void tout_kernel(const float* __restrict__ fused,
                                                   float* __restrict__ dst) {
    int idx = blockIdx.x * 256 + threadIdx.x;              // ordered (b,f,t)
    if (idx >= 16 * 256 * 1024) return;
    int t = idx & 1023;
    int f = (idx >> 10) & 255;
    int b = idx >> 18;
    dst[idx] = fused[(((size_t)b << 10) + t) * 256 + f];
}

// ---------------------------------------------------------------------------
// Tiled WMMA GEMM:  C[M,N] (f32) = A[M,K](bf16) * W[N,K]^T(bf16) + bias
// block tile 64x64, 8 waves, K-step 32, double-buffered async LDS staging.
// mode==1: head store into [B,C,T].  M must be a multiple of 64.
// ---------------------------------------------------------------------------
__global__ __launch_bounds__(256) void gemm_wmma_kernel(
    const bf16_t* __restrict__ A, const bf16_t* __restrict__ W,
    const float* __restrict__ bias, float* __restrict__ C,
    int M, int Nact, int K, int ldc, int mode) {
    __shared__ bf16_t As[2][64][40];   // 80-byte rows -> 16B aligned
    __shared__ bf16_t Bs[2][64][40];
    const int tid = threadIdx.x;
    const int lane = tid & 31, wave = tid >> 5;
    const int m0 = blockIdx.y * 64, n0 = blockIdx.x * 64;
    const int tm = wave & 3;            // M tile of this wave
    const int tnb = (wave >> 2) * 2;    // first of 2 N tiles
    v8f zero8 = {0.f, 0.f, 0.f, 0.f, 0.f, 0.f, 0.f, 0.f};
    v8f acc[2];
    acc[0] = zero8; acc[1] = zero8;

    const int lr = tid >> 2;            // 0..63  row within tile
    const int lc = (tid & 3) << 3;      // 0,8,16,24  k-chunk
    const int lm = lane & 15;

    // loop-invariant staging pointers
    const bf16_t* gA = A + (size_t)(m0 + lr) * K + lc;
    const bool bvalid = (n0 + lr) < Nact;
    const bf16_t* gB = W + (size_t)(bvalid ? (n0 + lr) : 0) * K + lc;

    // zero out-of-range B rows once (both buffers); async staging skips them
    if (!bvalid) {
        uint4 z = {0u, 0u, 0u, 0u};
        *(uint4*)&Bs[0][lr][lc] = z;
        *(uint4*)&Bs[1][lr][lc] = z;
    }

    const int nk = K >> 5;
    // prologue: stage tile 0 into buffer 0
    stage16(gA, &As[0][lr][lc]);
    if (bvalid) stage16(gB, &Bs[0][lr][lc]);
    stage_wait();
    __syncthreads();

    for (int i = 0; i < nk; ++i) {
        const int cur = i & 1, nxt = cur ^ 1;
        if (i + 1 < nk) {   // stage next tile while computing current
            const int ko = (i + 1) << 5;
            stage16(gA + ko, &As[nxt][lr][lc]);
            if (bvalid) stage16(gB + ko, &Bs[nxt][lr][lc]);
        }
        v16bf af = load_frag_a(&As[cur][tm * 16 + lm][0], lane);
#pragma unroll
        for (int j = 0; j < 2; ++j) {
            v16bf bfv = load_frag_b(&Bs[cur][(tnb + j) * 16 + lm][0], lane);
            acc[j] = __builtin_amdgcn_wmma_f32_16x16x32_bf16(
                false, af, false, bfv, (short)0, acc[j], false, false);
        }
        stage_wait();
        __syncthreads();
    }

    const int mrow = (lane >> 4) << 3;
#pragma unroll
    for (int j = 0; j < 2; ++j) {
        int tn = tnb + j;
        int n = n0 + tn * 16 + lm;
        float bvs = (bias != nullptr && n < Nact) ? bias[n] : 0.0f;
#pragma unroll
        for (int v = 0; v < 8; ++v) {
            int m = m0 + tm * 16 + mrow + v;
            if (n < Nact) {
                float val = acc[j][v] + bvs;
                if (mode == 0) {
                    C[(size_t)m * ldc + n] = val;
                } else {   // head: out[b, n, t] ; m = b*1024 + t
                    int b = m >> 10, t = m & 1023;
                    C[(((size_t)b * Nact + n) << 10) + t] = val;
                }
            }
        }
    }
}

// ---------------------------------------------------------------------------
// Row LayerNorm (+optional residual, +optional exact GELU); emits f32 and/or
// bf16 (bf16 with row stride + column offset, to fuse the concat).
// ---------------------------------------------------------------------------
__global__ __launch_bounds__(256) void ln_kernel(
    const float* __restrict__ X, const float* __restrict__ res,
    const float* __restrict__ g, const float* __restrict__ be,
    float* __restrict__ outF, bf16_t* __restrict__ outB,
    int N, int ldB, int colOff, int doGelu) {
    int row = blockIdx.x, tid = threadIdx.x;
    __shared__ float sbuf[256], qbuf[256];
    const float* xr = X + (size_t)row * N;
    const float* rr = res ? res + (size_t)row * N : nullptr;
    float vals[2]; float s = 0.f, q = 0.f;
#pragma unroll
    for (int i = 0; i < 2; ++i) {
        int n = tid + i * 256;
        float v = 0.f;
        if (n < N) { v = xr[n]; if (rr) v += rr[n]; }
        vals[i] = v; s += v; q += v * v;
    }
    sbuf[tid] = s; qbuf[tid] = q;
    __syncthreads();
    for (int o = 128; o > 0; o >>= 1) {
        if (tid < o) { sbuf[tid] += sbuf[tid + o]; qbuf[tid] += qbuf[tid + o]; }
        __syncthreads();
    }
    float mean = sbuf[0] / N;
    float var  = qbuf[0] / N - mean * mean;
    float inv  = rsqrtf(var + 1e-5f);
#pragma unroll
    for (int i = 0; i < 2; ++i) {
        int n = tid + i * 256;
        if (n < N) {
            float y = (vals[i] - mean) * inv * g[n] + be[n];
            if (doGelu) y = gelu_exact(y);
            if (outF) outF[(size_t)row * N + n] = y;
            if (outB) outB[(size_t)row * ldB + colOff + n] = (bf16_t)y;
        }
    }
}

// ---------------------------------------------------------------------------
// Scan helper: C_lds[16][N] (+bias) = A1_lds[16][K1]*W1^T (+ A2_lds*W2^T)
// A in LDS (bf16, row stride = K), W in global bf16 [N][K].  16 waves.
// ---------------------------------------------------------------------------
__device__ __forceinline__ void gemm16(
    const bf16_t* A1, int K1, const bf16_t* __restrict__ W1,
    const bf16_t* A2, int K2, const bf16_t* __restrict__ W2,
    const float* __restrict__ bias, float* C_lds, int N, int wave, int lane) {
    const int lm = lane & 15;
    const int mrow = (lane >> 4) << 3;
    for (int nt = wave; nt < (N >> 4); nt += 16) {
        const int n = (nt << 4) + lm;
        float b0 = bias ? bias[n] : 0.0f;
        v8f acc = {b0, b0, b0, b0, b0, b0, b0, b0};
        {
            const bf16_t* arow = A1 + lm * K1;
            const bf16_t* brow = W1 + (size_t)n * K1;
            for (int k0 = 0; k0 < K1; k0 += 32) {
                v16bf af = load_frag_a(arow + k0, lane);
                v16bf bfv = load_frag_b(brow + k0, lane);
                acc = __builtin_amdgcn_wmma_f32_16x16x32_bf16(
                    false, af, false, bfv, (short)0, acc, false, false);
            }
        }
        if (A2 != nullptr) {
            const bf16_t* arow = A2 + lm * K2;
            const bf16_t* brow = W2 + (size_t)n * K2;
            for (int k0 = 0; k0 < K2; k0 += 32) {
                v16bf af = load_frag_a(arow + k0, lane);
                v16bf bfv = load_frag_b(brow + k0, lane);
                acc = __builtin_amdgcn_wmma_f32_16x16x32_bf16(
                    false, af, false, bfv, (short)0, acc, false, false);
            }
        }
#pragma unroll
        for (int v = 0; v < 8; ++v)
            C_lds[(mrow + v) * N + (nt << 4) + lm] = acc[v];
    }
}

// ---------------------------------------------------------------------------
// Sequential scan: GRUCell + summary-MLP + capped-gate long memory.
// One persistent workgroup, all state in LDS (~233 KB of the 320 KB WGP LDS).
// ---------------------------------------------------------------------------
#define SCAN_SMEM 233472

__global__ __launch_bounds__(512) void scan_kernel(
    const float* __restrict__ xgru,   // [B*T, 768] f32 (includes bih)
    const float* __restrict__ xsum,   // [B*T, 384] f32 (includes s1_b)
    const bf16_t* __restrict__ Whh, const float* __restrict__ bhh,
    const bf16_t* __restrict__ Ws1h,
    const float* __restrict__ s_g, const float* __restrict__ s_be,
    const bf16_t* __restrict__ Ws2, const float* __restrict__ s2_bias,
    const bf16_t* __restrict__ Wgi, const float* __restrict__ gi_bias,
    const bf16_t* __restrict__ Wgs,
    const bf16_t* __restrict__ Wui, const float* __restrict__ ui_bias,
    const bf16_t* __restrict__ Wus,
    bf16_t* __restrict__ short_out,   // [B*T, 256] bf16
    bf16_t* __restrict__ long_out,    // [B*T, 384] bf16
    int T) {
    extern __shared__ char smem[];
    bf16_t* hb  = (bf16_t*)(smem);             // 16x256 bf16   8192
    float*  hf  = (float*)(smem + 8192);       // 16x256 f32    16384
    float*  mf  = (float*)(smem + 24576);      // 16x384 f32    24576
    bf16_t* mb  = (bf16_t*)(smem + 49152);     // 16x384 bf16   12288
    float*  gh  = (float*)(smem + 61440);      // 16x768 f32    49152
    float*  sp  = (float*)(smem + 110592);     // 16x384 f32    24576
    bf16_t* sB  = (bf16_t*)(smem + 135168);    // 16x384 bf16   12288
    float*  s2f = (float*)(smem + 147456);     // 16x384 f32    24576
    bf16_t* s2B = (bf16_t*)(smem + 172032);    // 16x384 bf16   12288
    float*  gp  = (float*)(smem + 184320);     // 16x384 f32    24576
    float*  pp  = (float*)(smem + 208896);     // 16x384 f32    24576

    const int tid = threadIdx.x;
    const int lane = tid & 31, wave = tid >> 5;   // 16 waves

    for (int i = tid; i < 16 * 256; i += 512) { hf[i] = 0.f; hb[i] = (bf16_t)0.0f; }
    for (int i = tid; i < 16 * 384; i += 512) { mf[i] = 0.f; mb[i] = (bf16_t)0.0f; }
    __syncthreads();

    for (int t = 0; t < T; ++t) {
        // 1) gh = h @ Whh^T + bhh            [16,768]
        gemm16(hb, 256, Whh, nullptr, 0, nullptr, bhh, gh, 768, wave, lane);
        __syncthreads();

        // 2) GRU gates -> h_new; emit short_seq
        for (int idx = tid; idx < 16 * 256; idx += 512) {
            int b = idx >> 8, j = idx & 255;
            size_t r = (size_t)b * T + t;
            const float* xg = xgru + r * 768;
            float hr = gh[b * 768 + j];
            float hz = gh[b * 768 + 256 + j];
            float hn = gh[b * 768 + 512 + j];
            float rg = sigmoidf_(xg[j] + hr);
            float zg = sigmoidf_(xg[256 + j] + hz);
            float ng = tanhf(xg[512 + j] + rg * hn);
            float hnew = (1.0f - zg) * ng + zg * hf[idx];
            hf[idx] = hnew;
            hb[idx] = (bf16_t)hnew;
            short_out[r * 256 + j] = (bf16_t)hnew;
        }
        __syncthreads();

        // 3) sp = h_new @ Ws1h^T             [16,384]
        gemm16(hb, 256, Ws1h, nullptr, 0, nullptr, nullptr, sp, 384, wave, lane);
        __syncthreads();

        // 4) s = GELU(LN(xsum + sp))  (one wave per batch row)
        {
            int b = wave;   // 16 waves, 16 rows
            const float* xs = xsum + ((size_t)b * T + t) * 384;
            float s = 0.f, q = 0.f;
            for (int j = lane; j < 384; j += 32) {
                float x = sp[b * 384 + j] + xs[j];
                sp[b * 384 + j] = x;
                s += x; q += x * x;
            }
            for (int o = 16; o > 0; o >>= 1) {
                s += __shfl_xor(s, o, 32);
                q += __shfl_xor(q, o, 32);
            }
            float mean = s / 384.0f;
            float var  = q / 384.0f - mean * mean;
            float inv  = rsqrtf(var + 1e-5f);
            for (int j = lane; j < 384; j += 32) {
                float y = (sp[b * 384 + j] - mean) * inv * s_g[j] + s_be[j];
                sB[b * 384 + j] = (bf16_t)gelu_exact(y);
            }
        }
        __syncthreads();

        // 5) s2 = s @ s2_W^T + s2_b
        gemm16(sB, 384, Ws2, nullptr, 0, nullptr, s2_bias, s2f, 384, wave, lane);
        __syncthreads();
        for (int idx = tid; idx < 16 * 384; idx += 512) s2B[idx] = (bf16_t)s2f[idx];
        __syncthreads();

        // 6) gate/prop pre-activations (two fused dual-K GEMMs)
        gemm16(s2B, 384, Wgi, mb, 384, Wgs, gi_bias, gp, 384, wave, lane);
        gemm16(s2B, 384, Wui, mb, 384, Wus, ui_bias, pp, 384, wave, lane);
        __syncthreads();

        // 7) m_new = (1-g)*m + g*tanh(prop); emit long_seq
        for (int idx = tid; idx < 16 * 384; idx += 512) {
            int b = idx / 384, j = idx - b * 384;
            float gate = 0.2f * sigmoidf_(gp[idx]);
            float prop = tanhf(pp[idx]);
            float mn = (1.0f - gate) * mf[idx] + gate * prop;
            mf[idx] = mn;
            mb[idx] = (bf16_t)mn;
            long_out[((size_t)b * T + t) * 384 + j] = (bf16_t)mn;
        }
        __syncthreads();
    }
}

// ---------------------------------------------------------------------------
extern "C" void kernel_launch(void* const* d_in, const int* in_sizes, int n_in,
                              void* d_out, int out_size, void* d_ws, size_t ws_size,
                              hipStream_t stream) {
    (void)in_sizes; (void)n_in; (void)out_size; (void)ws_size;
    enum { FEAT = 0, BP_W, BP_B, BP_G, BP_BE, GRU_WIH, GRU_BIH, GRU_WHH, GRU_BHH,
           S1_W, S1_B, S_G, S_BE, S2_W, S2_B, GI_W, GI_B, GS_W, UI_W, UI_B, US_W,
           BF_W, BF_B, BF_G, BF_BE, SF_W, SF_B, SF_G, SF_BE, LF_W, LF_B, LF_G, LF_BE,
           F1_W, F1_B, F_G, F_BE, F2_W, F2_B, FN_G, FN_BE, HEAD_W, HEAD_B };
    auto F = [&](int i) { return (const float*)d_in[i]; };

    const int Bb = 16, T = 1024, D = 1024, FD = 256, HID = 512, Cn = 200;
    const int M = Bb * T;  // 16384

    char* base_ws = (char*)d_ws;
    size_t off = 0;
    auto alloc = [&](size_t bytes) -> void* {
        void* p = base_ws + off;
        off = (off + bytes + 255) & ~(size_t)255;
        return p;
    };

    // bf16 weights
    bf16_t* wbp  = (bf16_t*)alloc((size_t)FD * D * 2);
    bf16_t* wgih = (bf16_t*)alloc((size_t)768 * 256 * 2);
    bf16_t* wghh = (bf16_t*)alloc((size_t)768 * 256 * 2);
    bf16_t* ws1x = (bf16_t*)alloc((size_t)384 * 256 * 2);
    bf16_t* ws1h = (bf16_t*)alloc((size_t)384 * 256 * 2);
    bf16_t* ws2  = (bf16_t*)alloc((size_t)384 * 384 * 2);
    bf16_t* wgi  = (bf16_t*)alloc((size_t)384 * 384 * 2);
    bf16_t* wgs  = (bf16_t*)alloc((size_t)384 * 384 * 2);
    bf16_t* wui  = (bf16_t*)alloc((size_t)384 * 384 * 2);
    bf16_t* wus  = (bf16_t*)alloc((size_t)384 * 384 * 2);
    bf16_t* wbf  = (bf16_t*)alloc((size_t)256 * 256 * 2);
    bf16_t* wsf  = (bf16_t*)alloc((size_t)256 * 256 * 2);
    bf16_t* wlf  = (bf16_t*)alloc((size_t)256 * 384 * 2);
    bf16_t* wf1  = (bf16_t*)alloc((size_t)512 * 768 * 2);
    bf16_t* wf2  = (bf16_t*)alloc((size_t)256 * 512 * 2);
    bf16_t* whead= (bf16_t*)alloc((size_t)200 * 256 * 2);

    // activations
    bf16_t* Xbf      = (bf16_t*)alloc((size_t)M * D * 2);
    float*  base_pre = (float*)alloc((size_t)M * FD * 4);
    bf16_t* base_bf  = (bf16_t*)alloc((size_t)M * FD * 2);
    float*  xgru     = (float*)alloc((size_t)M * 768 * 4);
    float*  xsum     = (float*)alloc((size_t)M * 384 * 4);
    bf16_t* short_bf = (bf16_t*)alloc((size_t)M * 256 * 2);
    bf16_t* long_bf  = (bf16_t*)alloc((size_t)M * 384 * 2);
    float*  pre      = (float*)alloc((size_t)M * 512 * 4);   // reused scratch
    float*  base_f   = (float*)alloc((size_t)M * 256 * 4);
    bf16_t* fin      = (bf16_t*)alloc((size_t)M * 768 * 2);
    bf16_t* dbf      = (bf16_t*)alloc((size_t)M * 512 * 2);
    float*  d2       = (float*)alloc((size_t)M * 256 * 4);
    float*  fused    = (float*)alloc((size_t)M * 256 * 4);
    bf16_t* fused_bf = (bf16_t*)alloc((size_t)M * 256 * 2);

    auto cvt = [&](const float* src, bf16_t* dst, int rows, int cols, int ld, int o) {
        int n = rows * cols;
        cvt_kernel<<<(n + 255) / 256, 256, 0, stream>>>(src, dst, rows, cols, ld, o);
    };
    auto gemm = [&](const bf16_t* A, const bf16_t* W, const float* bias, float* Cp,
                    int Mm, int Nact, int K, int ldc, int mode) {
        dim3 g((Nact + 63) / 64, Mm / 64);
        gemm_wmma_kernel<<<g, 256, 0, stream>>>(A, W, bias, Cp, Mm, Nact, K, ldc, mode);
    };
    auto ln = [&](const float* X, const float* res, const float* g, const float* be,
                  float* oF, bf16_t* oB, int N, int ldB, int co, int gelu) {
        ln_kernel<<<M, 256, 0, stream>>>(X, res, g, be, oF, oB, N, ldB, co, gelu);
    };

    // ---- weight conversion ----
    cvt(F(BP_W), wbp, FD, D, D, 0);
    cvt(F(GRU_WIH), wgih, 768, 256, 256, 0);
    cvt(F(GRU_WHH), wghh, 768, 256, 256, 0);
    cvt(F(S1_W), ws1x, 384, 256, 512, 0);
    cvt(F(S1_W), ws1h, 384, 256, 512, 256);
    cvt(F(S2_W), ws2, 384, 384, 384, 0);
    cvt(F(GI_W), wgi, 384, 384, 384, 0);
    cvt(F(GS_W), wgs, 384, 384, 384, 0);
    cvt(F(UI_W), wui, 384, 384, 384, 0);
    cvt(F(US_W), wus, 384, 384, 384, 0);
    cvt(F(BF_W), wbf, 256, 256, 256, 0);
    cvt(F(SF_W), wsf, 256, 256, 256, 0);
    cvt(F(LF_W), wlf, 256, 384, 384, 0);
    cvt(F(F1_W), wf1, 512, 768, 768, 0);
    cvt(F(F2_W), wf2, 256, 512, 512, 0);
    cvt(F(HEAD_W), whead, 200, 256, 256, 0);

    // ---- input transpose ----
    {
        size_t n = (size_t)16 * 1024 * 1024;
        tin_kernel<<<(unsigned)((n + 255) / 256), 256, 0, stream>>>(F(FEAT), Xbf);
    }

    // ---- base projector: GEMM + LN + GELU -> base (bf16) ----
    gemm(Xbf, wbp, F(BP_B), base_pre, M, FD, D, FD, 0);
    ln(base_pre, nullptr, F(BP_G), F(BP_BE), nullptr, base_bf, FD, FD, 0, 1);

    // ---- scan input projections ----
    gemm(base_bf, wgih, F(GRU_BIH), xgru, M, 768, 256, 768, 0);
    gemm(base_bf, ws1x, F(S1_B), xsum, M, 384, 256, 384, 0);

    // ---- sequential scan (persistent workgroup) ----
    scan_kernel<<<1, 512, SCAN_SMEM, stream>>>(
        xgru, xsum, wghh, F(GRU_BHH), ws1h, F(S_G), F(S_BE),
        ws2, F(S2_B), wgi, F(GI_B), wgs, wui, F(UI_B), wus,
        short_bf, long_bf, T);

    // ---- fuse projections (bf16 written straight into concat buffer) ----
    gemm(base_bf, wbf, F(BF_B), pre, M, 256, 256, 256, 0);
    ln(pre, nullptr, F(BF_G), F(BF_BE), base_f, fin, 256, 768, 0, 1);
    gemm(short_bf, wsf, F(SF_B), pre, M, 256, 256, 256, 0);
    ln(pre, nullptr, F(SF_G), F(SF_BE), nullptr, fin, 256, 768, 256, 1);
    gemm(long_bf, wlf, F(LF_B), pre, M, 256, 384, 256, 0);
    ln(pre, nullptr, F(LF_G), F(LF_BE), nullptr, fin, 256, 768, 512, 1);

    // ---- fusion MLP ----
    gemm(fin, wf1, F(F1_B), pre, M, HID, 768, HID, 0);
    ln(pre, nullptr, F(F_G), F(F_BE), nullptr, dbf, HID, HID, 0, 1);
    gemm(dbf, wf2, F(F2_B), d2, M, 256, HID, 256, 0);

    // ---- fused = LN(base_f + d) ----
    ln(d2, base_f, F(FN_G), F(FN_BE), fused, fused_bf, 256, 256, 0, 0);

    // ---- outputs ----
    float* out = (float*)d_out;
    float* out_logits = out;                               // [B, C, T]
    float* out_frames = out + (size_t)Bb * Cn * T;         // [B, FD, T]
    gemm(fused_bf, whead, F(HEAD_B), out_logits, M, Cn, 256, Cn, 1);
    tout_kernel<<<(16 * 256 * 1024) / 256, 256, 0, stream>>>(fused, out_frames);
}
